// Anotator_att_88115549044965
// MI455X (gfx1250) — compile-verified, matrix operand order checked by reading, not run on hardware
//
#include <hip/hip_runtime.h>
#include <math.h>

// ---------------- problem dimensions ----------------
#define B_    256
#define P_    196
#define E_    2048
#define H_    2048
#define A_    512
#define EMB_  300
#define VOUT_ 261
#define VOUTP_ 384            // fc rows padded to 3 * 128 (N-tile multiple)
#define T_    7
#define XK_   (EMB_ + E_)     // 2348 (raw x width)
#define XKP_  2368            // x width padded: multiple of 32, rows 16B-aligned
#define KCAT_ (XKP_ + H_)     // 4416 = concat [x | h] K for fused LSTM GEMM
#define G4H_  (4 * H_)        // 8192

// ---------------- WMMA types ----------------
typedef __attribute__((ext_vector_type(16))) __bf16 v16bf;
typedef __attribute__((ext_vector_type(8)))  float  v8f;

// Unguarded 16-row bf16 fragment load (ISA 16-bit A-matrix 16x32 layout):
// two 16B loads, zero VALU. p already points at (row, khalf) of the tile.
__device__ __forceinline__ v16bf ldfrag(const __bf16* __restrict__ p, int k) {
  union { uint4 q[2]; v16bf bf; } u;
  u.q[0] = *(const uint4*)(p + k);
  u.q[1] = *(const uint4*)(p + k + 16);
  return u.bf;
}

#define WMMA8()                                                                                              \
  do {                                                                                                       \
    acc[0][0] = __builtin_amdgcn_wmma_f32_16x16x32_bf16(false, a0, false, b0, (short)0, acc[0][0], false, false); \
    acc[0][1] = __builtin_amdgcn_wmma_f32_16x16x32_bf16(false, a0, false, b1, (short)0, acc[0][1], false, false); \
    acc[0][2] = __builtin_amdgcn_wmma_f32_16x16x32_bf16(false, a0, false, b2, (short)0, acc[0][2], false, false); \
    acc[0][3] = __builtin_amdgcn_wmma_f32_16x16x32_bf16(false, a0, false, b3, (short)0, acc[0][3], false, false); \
    acc[1][0] = __builtin_amdgcn_wmma_f32_16x16x32_bf16(false, a1, false, b0, (short)0, acc[1][0], false, false); \
    acc[1][1] = __builtin_amdgcn_wmma_f32_16x16x32_bf16(false, a1, false, b1, (short)0, acc[1][1], false, false); \
    acc[1][2] = __builtin_amdgcn_wmma_f32_16x16x32_bf16(false, a1, false, b2, (short)0, acc[1][2], false, false); \
    acc[1][3] = __builtin_amdgcn_wmma_f32_16x16x32_bf16(false, a1, false, b3, (short)0, acc[1][3], false, false); \
  } while (0)

// ---------------- WMMA GEMM:  C = act(A (MxK,bf16) * W(NxK,bf16)^T + bias) ----
// ACT: 0 none, 1 sigmoid.  OUTBF: bf16 output.  NG: store-side N guard (yhat).
// block = 256 thr = 8 waves (4 along M x 2 along N); block tile 128x128;
// wave tile 32(M) x 64(N) = 2x4 WMMA accumulators.
// Requirements: K % 32 == 0; M % 128 == 0; operand buffers cover the full
// (padded) tile range so loads never go out of bounds.
template <int ACT, bool OUTBF, bool NG>
__global__ __launch_bounds__(256) void gemm_bf16_wmma(
    const __bf16* __restrict__ A, int lda,
    const __bf16* __restrict__ W, int ldw,
    const float* __restrict__ bias,
    void* __restrict__ Cv, int ldc,
    int M, int N, int K)
{
  const int lane  = threadIdx.x & 31;
  const int wave  = threadIdx.x >> 5;
  const int wm    = wave >> 1;            // 0..3
  const int wn    = wave & 1;             // 0..1
  const int m0    = blockIdx.y * 128 + wm * 32;
  const int n0    = blockIdx.x * 128 + wn * 64;
  const int row   = lane & 15;
  const int khalf = (lane >> 4) << 3;     // 0 or 8
  const int mofs  = (lane >> 4) << 3;     // C/D: lanes 16-31 hold M+8

  const __bf16* pa0 = A + (size_t)(m0 + row) * lda + khalf;
  const __bf16* pa1 = pa0 + (size_t)16 * lda;
  const __bf16* pb0 = W + (size_t)(n0 + row) * ldw + khalf;
  const __bf16* pb1 = pb0 + (size_t)16 * ldw;
  const __bf16* pb2 = pb0 + (size_t)32 * ldw;
  const __bf16* pb3 = pb0 + (size_t)48 * ldw;

  v8f acc[2][4];
#pragma unroll
  for (int i = 0; i < 2; ++i)
#pragma unroll
    for (int j = 0; j < 4; ++j)
#pragma unroll
      for (int v = 0; v < 8; ++v) acc[i][j][v] = 0.f;

  // software-pipelined K loop: loads for k+32 overlap WMMAs on k
  v16bf a0 = ldfrag(pa0, 0), a1 = ldfrag(pa1, 0);
  v16bf b0 = ldfrag(pb0, 0), b1 = ldfrag(pb1, 0);
  v16bf b2 = ldfrag(pb2, 0), b3 = ldfrag(pb3, 0);
  for (int k0 = 32; k0 < K; k0 += 32) {
    v16bf na0 = ldfrag(pa0, k0), na1 = ldfrag(pa1, k0);
    v16bf nb0 = ldfrag(pb0, k0), nb1 = ldfrag(pb1, k0);
    v16bf nb2 = ldfrag(pb2, k0), nb3 = ldfrag(pb3, k0);
    WMMA8();
    a0 = na0; a1 = na1; b0 = nb0; b1 = nb1; b2 = nb2; b3 = nb3;
  }
  WMMA8();

#pragma unroll
  for (int i = 0; i < 2; ++i)
#pragma unroll
    for (int j = 0; j < 4; ++j)
#pragma unroll
      for (int v = 0; v < 8; ++v) {
        int m = m0 + i * 16 + v + mofs;
        int n = n0 + j * 16 + (lane & 15);
        if (!NG || n < N) {
          float val = acc[i][j][v] + bias[n];
          if (ACT == 1) val = 1.f / (1.f + expf(-val));
          if (OUTBF) ((__bf16*)Cv)[(size_t)m * ldc + n] = (__bf16)val;
          else       ((float*)Cv)[(size_t)m * ldc + n] = val;
        }
      }
}

// ---------------- setup / conversion kernels ----------------
__global__ __launch_bounds__(256) void zero_kernel(float* __restrict__ p, int n) {
  int i = blockIdx.x * blockDim.x + threadIdx.x;
  if (i < n) p[i] = 0.f;
}

// fp32 -> bf16, 4 elements per thread (n must be a multiple of 4)
__global__ __launch_bounds__(256) void cvt4_kernel(const float* __restrict__ s,
                                                   __bf16* __restrict__ d, long n4) {
  long i = (long)blockIdx.x * 256 + threadIdx.x;
  if (i >= n4) return;
  float4 v = ((const float4*)s)[i];
  union { __bf16 h[4]; uint2 u; } o;
  o.h[0] = (__bf16)v.x; o.h[1] = (__bf16)v.y;
  o.h[2] = (__bf16)v.z; o.h[3] = (__bf16)v.w;
  ((uint2*)d)[i] = o.u;
}

// Wcat[r, 0:2348)=W_ih[r], [2348:2368)=0, [2368:4416)=W_hh[r]   (bf16)
__global__ __launch_bounds__(256) void pack_wcat_kernel(
    const float* __restrict__ W_ih, const float* __restrict__ W_hh,
    __bf16* __restrict__ Wcat) {
  int r = blockIdx.x;
  __bf16* d = Wcat + (size_t)r * KCAT_;
  for (int c = threadIdx.x; c < XKP_; c += 256)
    d[c] = (c < XK_) ? (__bf16)W_ih[(size_t)r * XK_ + c] : (__bf16)0.f;
  for (int c = threadIdx.x; c < H_; c += 256)
    d[XKP_ + c] = (__bf16)W_hh[(size_t)r * H_ + c];
}

// fc weights padded to VOUTP_ zero-filled rows; bias padded too
__global__ __launch_bounds__(256) void pack_fc_kernel(
    const float* __restrict__ fc_W, const float* __restrict__ fc_b,
    __bf16* __restrict__ fcp, float* __restrict__ fcbp) {
  int r = blockIdx.x;          // 0..VOUTP_-1
  __bf16* d = fcp + (size_t)r * H_;
  if (r < VOUT_) {
    for (int c = threadIdx.x; c < H_; c += 256) d[c] = (__bf16)fc_W[(size_t)r * H_ + c];
  } else {
    for (int c = threadIdx.x; c < H_; c += 256) d[c] = (__bf16)0.f;
  }
  if (threadIdx.x == 0) fcbp[r] = (r < VOUT_) ? fc_b[r] : 0.f;
}

__global__ __launch_bounds__(256) void bias_sum_kernel(
    const float* __restrict__ a, const float* __restrict__ b,
    float* __restrict__ o, int n) {
  int i = blockIdx.x * blockDim.x + threadIdx.x;
  if (i < n) o[i] = a[i] + b[i];
}

// zero the static pad columns of xh (cols XK_..XKP_)
__global__ void xh_pad_kernel(__bf16* __restrict__ xh) {
  int b = blockIdx.x;
  if (threadIdx.x < (XKP_ - XK_))
    xh[(size_t)b * KCAT_ + XK_ + threadIdx.x] = (__bf16)0.f;
}

__global__ __launch_bounds__(256) void h_to_xh_kernel(const __bf16* __restrict__ hb,
                                                      __bf16* __restrict__ xh) {
  int b = blockIdx.x;
  for (int j = threadIdx.x; j < H_; j += 256)
    xh[(size_t)b * KCAT_ + XKP_ + j] = hb[(size_t)b * H_ + j];
}

// ---------------- attention score ----------------
// e[b,p] = relu(att1[b,p,:] + att2[b,:]) . full_w + full_b ; one wave per (b,p)
__global__ __launch_bounds__(256) void att_score_kernel(
    const __bf16* __restrict__ att1, const float* __restrict__ att2,
    const float* __restrict__ full_w, const float* __restrict__ full_b,
    float* __restrict__ e)
{
  int gw = blockIdx.x * 8 + (threadIdx.x >> 5);
  if (gw >= B_ * P_) return;
  int lane = threadIdx.x & 31;
  int b = gw / P_;
  const __bf16* a1 = att1 + (size_t)gw * A_;
  const float*  a2 = att2 + (size_t)b * A_;
  float s = 0.f;
#pragma unroll 4
  for (int a = lane; a < A_; a += 32) {
    float v = (float)a1[a] + a2[a];
    v = v > 0.f ? v : 0.f;
    s += v * full_w[a];
  }
  for (int off = 16; off; off >>= 1) s += __shfl_down(s, off, 32);
  if (lane == 0) e[gw] = s + full_b[0];
}

// ---------------- softmax over P per batch ----------------
__global__ __launch_bounds__(256) void softmax_kernel(
    const float* __restrict__ e, float* __restrict__ alpha_ws,
    float* __restrict__ alphas_out, int t)
{
  __shared__ float sm[8];
  int b = blockIdx.x;
  int tid = threadIdx.x, lane = tid & 31, wv = tid >> 5;
  float v = (tid < P_) ? e[b * P_ + tid] : -3.4e38f;
  float m = v;
  for (int off = 16; off; off >>= 1) m = fmaxf(m, __shfl_down(m, off, 32));
  if (lane == 0) sm[wv] = m;
  __syncthreads();
  if (wv == 0) {
    float mm = (lane < 8) ? sm[lane] : -3.4e38f;
    for (int off = 4; off; off >>= 1) mm = fmaxf(mm, __shfl_down(mm, off, 32));
    if (lane == 0) sm[0] = mm;
  }
  __syncthreads();
  float mx = sm[0];
  __syncthreads();
  float ex = (tid < P_) ? expf(v - mx) : 0.f;
  float s = ex;
  for (int off = 16; off; off >>= 1) s += __shfl_down(s, off, 32);
  if (lane == 0) sm[wv] = s;
  __syncthreads();
  if (wv == 0) {
    float ss = (lane < 8) ? sm[lane] : 0.f;
    for (int off = 4; off; off >>= 1) ss += __shfl_down(ss, off, 32);
    if (lane == 0) sm[0] = ss;
  }
  __syncthreads();
  float inv = 1.f / sm[0];
  if (tid < P_) {
    float a = ex * inv;
    alpha_ws[b * P_ + tid] = a;
    alphas_out[(size_t)b * (T_ * P_) + t * P_ + tid] = a;
  }
}

// ---------------- awe[b,:] = sum_p alpha[b,p] * enc_bf[b,p,:] ----------------
__global__ __launch_bounds__(256) void awe_kernel(
    const float* __restrict__ alpha, const __bf16* __restrict__ enc,
    float* __restrict__ awe)
{
  int b = blockIdx.x, tid = threadIdx.x;
  const __bf16* encb = enc + (size_t)b * P_ * E_;
  const float* al = alpha + b * P_;
  float acc[E_ / 256];
#pragma unroll
  for (int j = 0; j < E_ / 256; ++j) acc[j] = 0.f;
  for (int p = 0; p < P_; ++p) {
    float w = al[p];
    const __bf16* r = encb + (size_t)p * E_;
#pragma unroll
    for (int j = 0; j < E_ / 256; ++j) acc[j] += w * (float)r[tid + j * 256];
  }
#pragma unroll
  for (int j = 0; j < E_ / 256; ++j) awe[(size_t)b * E_ + tid + j * 256] = acc[j];
}

// ---------------- x-part of xh = [emb_W[token], gate*awe] (bf16) -------------
__global__ __launch_bounds__(256) void build_x_kernel(
    const int* __restrict__ ann, int t, const float* __restrict__ emb_W,
    const float* __restrict__ gate, const float* __restrict__ awe,
    __bf16* __restrict__ xh)
{
  int b = blockIdx.x;
  int token = ann[b * T_ + t];
  __bf16* xr = xh + (size_t)b * KCAT_;
  for (int i = threadIdx.x; i < EMB_; i += blockDim.x)
    xr[i] = (__bf16)emb_W[(size_t)token * EMB_ + i];
  for (int i = threadIdx.x; i < E_; i += blockDim.x)
    xr[EMB_ + i] = (__bf16)(gate[(size_t)b * E_ + i] * awe[(size_t)b * E_ + i]);
}

// ---------------- LSTM pointwise: writes c(f32), h(bf16) + xh h-section ------
__global__ __launch_bounds__(256) void lstm_cell_kernel(
    const float* __restrict__ gates, float* __restrict__ c,
    __bf16* __restrict__ hb, __bf16* __restrict__ xh)
{
  int idx = blockIdx.x * blockDim.x + threadIdx.x;
  if (idx >= B_ * H_) return;
  int b = idx / H_, j = idx - b * H_;
  const float* g = gates + (size_t)b * G4H_;
  float ig = 1.f / (1.f + expf(-g[j]));
  float fg = 1.f / (1.f + expf(-g[H_ + j]));
  float gg = tanhf(g[2 * H_ + j]);
  float og = 1.f / (1.f + expf(-g[3 * H_ + j]));
  float cn = fg * c[idx] + ig * gg;
  c[idx] = cn;
  __bf16 hn = (__bf16)(og * tanhf(cn));
  hb[idx] = hn;
  xh[(size_t)b * KCAT_ + XKP_ + j] = hn;
}

// ---------------- host-side orchestration ----------------
extern "C" void kernel_launch(void* const* d_in, const int* in_sizes, int n_in,
                              void* d_out, int out_size, void* d_ws, size_t ws_size,
                              hipStream_t stream) {
  const float* enc       = (const float*)d_in[0];   // (B,P,E)
  const float* enc_out   = (const float*)d_in[1];   // (B,E)
  const int*   ann       = (const int*)  d_in[2];   // (B,T)
  const float* emb_W     = (const float*)d_in[4];   // (V_IN,EMB)
  const float* feat_W    = (const float*)d_in[5];   // (H,E)
  const float* feat_b    = (const float*)d_in[6];
  const float* W_ih      = (const float*)d_in[7];   // (4H, 2348)
  const float* W_hh      = (const float*)d_in[8];   // (4H, H)
  const float* b_ih      = (const float*)d_in[9];
  const float* b_hh      = (const float*)d_in[10];
  const float* enc_att_W = (const float*)d_in[11];  // (A,E)
  const float* enc_att_b = (const float*)d_in[12];
  const float* dec_att_W = (const float*)d_in[13];  // (A,H)
  const float* dec_att_b = (const float*)d_in[14];
  const float* full_W    = (const float*)d_in[15];  // (1,A)
  const float* full_b    = (const float*)d_in[16];  // (1,)
  const float* fbeta_W   = (const float*)d_in[17];  // (E,H)
  const float* fbeta_b   = (const float*)d_in[18];
  const float* fc_W      = (const float*)d_in[19];  // (V_OUT,H)
  const float* fc_b      = (const float*)d_in[20];
  (void)in_sizes; (void)n_in; (void)out_size; (void)ws_size;

  float* out = (float*)d_out;
  float* yhats  = out;                              // (B,T,V_OUT)
  float* alphas = out + (size_t)B_ * T_ * VOUT_;    // (B,T,P)

  // ---- workspace layout ----
  char* w = (char*)d_ws;
  size_t o = 0;
  auto alloc = [&](size_t bytes) { char* p = w + o; o = (o + bytes + 255) & ~(size_t)255; return p; };
  __bf16* enc_bf     = (__bf16*)alloc((size_t)B_ * P_ * E_ * 2);
  __bf16* att1_bf    = (__bf16*)alloc((size_t)B_ * P_ * A_ * 2);
  __bf16* Wcat_bf    = (__bf16*)alloc((size_t)G4H_ * KCAT_ * 2);
  __bf16* enc_out_bf = (__bf16*)alloc((size_t)B_ * E_ * 2);
  __bf16* enc_att_bf = (__bf16*)alloc((size_t)A_ * E_ * 2);
  __bf16* feat_bf    = (__bf16*)alloc((size_t)H_ * E_ * 2);
  __bf16* dec_att_bf = (__bf16*)alloc((size_t)A_ * H_ * 2);
  __bf16* fbeta_bf   = (__bf16*)alloc((size_t)E_ * H_ * 2);
  __bf16* fc_bf      = (__bf16*)alloc((size_t)VOUTP_ * H_ * 2);
  __bf16* h_bf       = (__bf16*)alloc((size_t)B_ * H_ * 2);
  __bf16* xh         = (__bf16*)alloc((size_t)B_ * KCAT_ * 2);
  float*  c          = (float*) alloc((size_t)B_ * H_ * 4);
  float*  att2       = (float*) alloc((size_t)B_ * A_ * 4);
  float*  ebuf       = (float*) alloc((size_t)B_ * P_ * 4);
  float*  alpha      = (float*) alloc((size_t)B_ * P_ * 4);
  float*  awe        = (float*) alloc((size_t)B_ * E_ * 4);
  float*  gate       = (float*) alloc((size_t)B_ * E_ * 4);
  float*  gates      = (float*) alloc((size_t)B_ * G4H_ * 4);
  float*  bsum       = (float*) alloc((size_t)G4H_ * 4);
  float*  fcb_pad    = (float*) alloc((size_t)VOUTP_ * 4);

  // ---- one-time setup (per launch; deterministic) ----
  zero_kernel<<<(B_ * H_ + 255) / 256, 256, 0, stream>>>(c, B_ * H_);
  cvt4_kernel<<<(int)(((long)B_ * P_ * E_ / 4 + 255) / 256), 256, 0, stream>>>(enc, enc_bf, (long)B_ * P_ * E_ / 4);
  cvt4_kernel<<<(B_ * E_ / 4 + 255) / 256, 256, 0, stream>>>(enc_out, enc_out_bf, B_ * E_ / 4);
  cvt4_kernel<<<(A_ * E_ / 4 + 255) / 256, 256, 0, stream>>>(enc_att_W, enc_att_bf, A_ * E_ / 4);
  cvt4_kernel<<<(H_ * E_ / 4 + 255) / 256, 256, 0, stream>>>(feat_W, feat_bf, H_ * E_ / 4);
  cvt4_kernel<<<(A_ * H_ / 4 + 255) / 256, 256, 0, stream>>>(dec_att_W, dec_att_bf, A_ * H_ / 4);
  cvt4_kernel<<<(E_ * H_ / 4 + 255) / 256, 256, 0, stream>>>(fbeta_W, fbeta_bf, E_ * H_ / 4);
  pack_wcat_kernel<<<G4H_, 256, 0, stream>>>(W_ih, W_hh, Wcat_bf);
  pack_fc_kernel<<<VOUTP_, 256, 0, stream>>>(fc_W, fc_b, fc_bf, fcb_pad);
  bias_sum_kernel<<<(G4H_ + 255) / 256, 256, 0, stream>>>(b_ih, b_hh, bsum, G4H_);
  xh_pad_kernel<<<B_, 32, 0, stream>>>(xh);

  // h0 = enc_out @ feat_W^T + feat_b  (bf16 out)   M=256 N=2048 K=2048
  gemm_bf16_wmma<0, true, false><<<dim3(H_ / 128, B_ / 128), 256, 0, stream>>>(
      enc_out_bf, E_, feat_bf, E_, feat_b, h_bf, H_, B_, H_, E_);
  h_to_xh_kernel<<<B_, 256, 0, stream>>>(h_bf, xh);

  // att1 = enc @ enc_att_W^T + enc_att_b (bf16 out)  M=50176 N=512 K=2048
  gemm_bf16_wmma<0, true, false><<<dim3(A_ / 128, (B_ * P_) / 128), 256, 0, stream>>>(
      enc_bf, E_, enc_att_bf, E_, enc_att_b, att1_bf, A_, B_ * P_, A_, E_);

  for (int t = 0; t < T_; ++t) {
    // att2 = h @ dec_att_W^T + dec_att_b   M=256 N=512 K=2048
    gemm_bf16_wmma<0, false, false><<<dim3(A_ / 128, B_ / 128), 256, 0, stream>>>(
        h_bf, H_, dec_att_bf, H_, dec_att_b, att2, A_, B_, A_, H_);

    att_score_kernel<<<(B_ * P_) / 8, 256, 0, stream>>>(att1_bf, att2, full_W, full_b, ebuf);
    softmax_kernel<<<B_, 256, 0, stream>>>(ebuf, alpha, alphas, t);
    awe_kernel<<<B_, 256, 0, stream>>>(alpha, enc_bf, awe);

    // gate = sigmoid(h @ fbeta_W^T + fbeta_b)   M=256 N=2048 K=2048
    gemm_bf16_wmma<1, false, false><<<dim3(E_ / 128, B_ / 128), 256, 0, stream>>>(
        h_bf, H_, fbeta_bf, H_, fbeta_b, gate, E_, B_, E_, H_);

    build_x_kernel<<<B_, 256, 0, stream>>>(ann, t, emb_W, gate, awe, xh);

    // gates = [x|h] @ [W_ih|W_hh]^T + (b_ih+b_hh)   M=256 N=8192 K=4416
    gemm_bf16_wmma<0, false, false><<<dim3(G4H_ / 128, B_ / 128), 256, 0, stream>>>(
        xh, KCAT_, Wcat_bf, KCAT_, bsum, gates, G4H_, B_, G4H_, KCAT_);

    lstm_cell_kernel<<<(B_ * H_ + 255) / 256, 256, 0, stream>>>(gates, c, h_bf, xh);

    // yhat = h @ fc_W(padded)^T + fc_b -> yhats[:, t, :]  M=256 N=261(384) K=2048
    gemm_bf16_wmma<0, false, true><<<dim3(VOUTP_ / 128, B_ / 128), 256, 0, stream>>>(
        h_bf, H_, fc_bf, H_, fcb_pad, yhats + (size_t)t * VOUT_, T_ * VOUT_, B_, VOUT_, H_);
  }
}
